// SSMBlock_89893665505537
// MI455X (gfx1250) — matrix operand order repaired and checked by reading, not compiled, not run on hardware
//
#include <hip/hip_runtime.h>
#include <hip/hip_bf16.h>
#include <math.h>

typedef _Float16 f16;
typedef __attribute__((ext_vector_type(16))) _Float16 v16h;
typedef __attribute__((ext_vector_type(8)))  _Float16 v8h;
typedef __attribute__((ext_vector_type(8)))  float    v8f;

#define B_   8
#define T_   2048
#define C_   512
#define L_   2048
#define CR_  32
#define NUP_ 1024
#define BT_  (B_*T_)
#define TP_  (T_ + 64)      /* x row stride in LDS: 64 zero halfwords of high pad */
#define KPAD_ 64            /* zero halfwords in front of reversed kernel */
#define EPS_ 1e-5f

#define CAT16(lo, hi) __builtin_shufflevector(lo, hi, 0,1,2,3,4,5,6,7,8,9,10,11,12,13,14,15)

// ---- CDNA5 async global->LDS copy (ASYNCcnt-tracked DMA path) --------------
__device__ __forceinline__ void async_ld_b128(void* lds, const void* g) {
  asm volatile("global_load_async_to_lds_b128 %0, %1, off"
               :: "v"((uint32_t)(uintptr_t)lds), "v"(g)
               : "memory");
}
__device__ __forceinline__ void wait_async0() {
  asm volatile("s_wait_asynccnt 0" ::: "memory");
}

// ---------------------------------------------------------------------------
// 0a) Pack x [B,T,C] f32 -> xT [C][B*T] f16, TIME-REVERSED within each row:
//     xT[c][b*T + (T-1-t)] = x[b,t,c]   (async-staged tiled transpose)
// ---------------------------------------------------------------------------
__global__ void pack_x_kernel(const float* __restrict__ x, f16* __restrict__ xT) {
  __shared__ __align__(16) float tile[32][36];
  int b  = blockIdx.z;
  int t0 = blockIdx.y * 32;
  int c0 = blockIdx.x * 32;
  int tx = threadIdx.x, ty = threadIdx.y;   // 32 x 8
  int tid = ty * 32 + tx;
  {
    int r = tid >> 3, ck = (tid & 7) * 4;   // 32 rows x 8 chunks of 4 floats
    async_ld_b128(&tile[r][ck], x + ((size_t)b * T_ + t0 + r) * C_ + c0 + ck);
  }
  wait_async0();
  __syncthreads();
#pragma unroll
  for (int i = 0; i < 4; i++) {
    int c = c0 + ty + i * 8;
    xT[(size_t)c * BT_ + (size_t)b * T_ + (T_ - 1 - (t0 + tx))] = (f16)tile[tx][ty + i * 8];
  }
}

// 0b) k [C,1,L] f32 -> krev [C][L] f16, krev[c][d] = k[c, L-1-d]
__global__ void krev_kernel(const float* __restrict__ k, f16* __restrict__ kr) {
  int idx = blockIdx.x * blockDim.x + threadIdx.x;
  if (idx >= C_ * L_) return;
  int c = idx >> 11, d = idx & (L_ - 1);
  kr[(size_t)c * L_ + d] = (f16)k[(size_t)c * L_ + (L_ - 1 - d)];
}

// 0c) Weights [K,N] f32 -> [N,K] f16 (so GEMM B-fragments are K-contiguous)
__global__ void cvt_w_transpose_kernel(const float* __restrict__ w, f16* __restrict__ wt,
                                       int K, int N) {
  __shared__ float tile[32][33];
  int n0 = blockIdx.x * 32, k0 = blockIdx.y * 32;
  int tx = threadIdx.x, ty = threadIdx.y;   // 32 x 8
#pragma unroll
  for (int i = 0; i < 4; i++)
    tile[ty + i * 8][tx] = w[(size_t)(k0 + ty + i * 8) * N + n0 + tx];
  __syncthreads();
#pragma unroll
  for (int i = 0; i < 4; i++)
    wt[(size_t)(n0 + ty + i * 8) * K + k0 + tx] = (f16)tile[tx][ty + i * 8];
}

// ---------------------------------------------------------------------------
// 1) Causal depthwise long conv as block-Toeplitz WMMA.
//    One block = one channel, 4 waves, pairs of output tiles (I, I+2) share A.
//    Channel staging (36 KB) done with async global->LDS b128 copies.
//    Output y stored as f16 (elementwise consumer only).
// ---------------------------------------------------------------------------
__global__ void __launch_bounds__(128) conv_kernel(const f16* __restrict__ xT,
                                                   const f16* __restrict__ kr,
                                                   f16* __restrict__ y,
                                                   float* __restrict__ ssum_g) {
  __shared__ __align__(16) f16 ksh[KPAD_ + L_];   // 4.2 KB
  __shared__ __align__(16) f16 xsh[B_ * TP_];     // 33.8 KB (rows high-padded w/ zeros)
  __shared__ float ssum[B_];
  int c   = blockIdx.x;
  int tid = threadIdx.x;
  // async DMA of the whole channel: x rows (2048 b128) + kernel (256 b128)
  for (int i = tid; i < 2048; i += 128) {
    int row = i >> 8, j = (i & 255) * 8;
    async_ld_b128(&xsh[row * TP_ + j], xT + (size_t)c * BT_ + row * T_ + j);
  }
  for (int i = tid; i < 256; i += 128)
    async_ld_b128(&ksh[KPAD_ + i * 8], kr + (size_t)c * L_ + i * 8);
  // zero pads (front of kernel, back of each x row)
  for (int i = tid; i < KPAD_; i += 128)      ksh[i] = (f16)0.0f;
  for (int i = tid; i < B_ * 64; i += 128)    xsh[(i >> 6) * TP_ + T_ + (i & 63)] = (f16)0.0f;
  if (tid < B_) ssum[tid] = 0.0f;
  wait_async0();
  __syncthreads();

  int wave = tid >> 5, lane = tid & 31;
  int li = lane & 15, kh = lane >> 4;
  int bcol = li & 7, gcol = li >> 3;     // column -> (batch, which tile of the pair)
  float lsum = 0.0f;

  for (int p = wave; p < 64; p += 4) {
    int I = ((p >> 1) << 2) + (p & 1);   // pairs (I, I+2): I in {0,1,4,5,8,9,...}
    v8f acc0 = {}, acc1 = {};
    int nch = (I + 4) >> 1;              // chunks for tile I+2; q = I+2-2*ch
    for (int ch = 0; ch < nch; ch++) {
      int q = I + 2 - 2 * ch;
      v16h a;
      int dbase = KPAD_ + 16 * q + li - 31 + kh * 16;
#pragma unroll
      for (int e = 0; e < 8; e++) {
        a[e]     = ksh[dbase + e];
        a[8 + e] = ksh[dbase + 8 + e];
      }
      int s0 = 32 * (ch - 1 + gcol);
      const v8h* bp = (const v8h*)&xsh[bcol * TP_ + (T_ - 32 - s0) + kh * 16];
      v16h bm = CAT16(bp[0], bp[1]);
      if (ch & 1)
        acc1 = __builtin_amdgcn_wmma_f32_16x16x32_f16(false, a, false, bm, (short)0, acc1, false, false);
      else
        acc0 = __builtin_amdgcn_wmma_f32_16x16x32_f16(false, a, false, bm, (short)0, acc0, false, false);
    }
    v8f acc = acc0 + acc1;
    int tbase = 16 * (I + 2 * gcol);
#pragma unroll
    for (int r = 0; r < 8; r++) {
      float v = acc[r];
      y[((size_t)bcol * T_ + tbase + r + kh * 8) * C_ + c] = (f16)v;
      lsum += v;
    }
  }
  atomicAdd(&ssum[bcol], lsum);
  __syncthreads();
  if (tid < B_) ssum_g[(size_t)tid * C_ + c] = ssum[tid];
}

// ---------------------------------------------------------------------------
// 2) SE gate: g = sigmoid(relu(mean_t(y) @ w1 + b1) @ w2 + b2).  Tiny: 1 block.
// ---------------------------------------------------------------------------
__global__ void se_kernel(const float* __restrict__ ssum, const float* __restrict__ w1,
                          const float* __restrict__ b1, const float* __restrict__ w2,
                          const float* __restrict__ b2, float* __restrict__ g) {
  __shared__ float hid[B_][CR_];
  int tid = threadIdx.x;                 // 256 = B_*CR_
  int b = tid >> 5, j = tid & 31;
  float acc = b1[j];
  const float inv = 1.0f / (float)T_;
  for (int cc = 0; cc < C_; cc++)
    acc += ssum[b * C_ + cc] * inv * w1[cc * CR_ + j];
  hid[b][j] = fmaxf(acc, 0.0f);
  __syncthreads();
  for (int idx = tid; idx < B_ * C_; idx += 256) {
    int bb = idx >> 9, cc = idx & (C_ - 1);
    float a = b2[cc];
#pragma unroll
    for (int jj = 0; jj < CR_; jj++) a += hid[bb][jj] * w2[jj * C_ + cc];
    g[idx] = 1.0f / (1.0f + expf(-a));
  }
}

// ---------------------------------------------------------------------------
// 3) y1 = LN(x + y*g) ; also f16 copy for the up-GEMM.  One block per row.
// ---------------------------------------------------------------------------
__global__ void __launch_bounds__(128) fuse1_kernel(const float* __restrict__ x,
                                                    const f16* __restrict__ y,
                                                    const float* __restrict__ g,
                                                    const float* __restrict__ w,
                                                    const float* __restrict__ bb,
                                                    float* __restrict__ y1,
                                                    f16* __restrict__ y1h) {
  __shared__ float r1[128], r2[128];
  int row = blockIdx.x;
  int b = row / T_;
  int tid = threadIdx.x;
  float v[4]; float sum = 0.0f, sq = 0.0f;
#pragma unroll
  for (int j = 0; j < 4; j++) {
    int cidx = tid + j * 128;
    float vv = x[(size_t)row * C_ + cidx] + (float)y[(size_t)row * C_ + cidx] * g[b * C_ + cidx];
    v[j] = vv; sum += vv; sq += vv * vv;
  }
  r1[tid] = sum; r2[tid] = sq; __syncthreads();
  for (int s = 64; s > 0; s >>= 1) {
    if (tid < s) { r1[tid] += r1[tid + s]; r2[tid] += r2[tid + s]; }
    __syncthreads();
  }
  float mean = r1[0] * (1.0f / C_);
  float var  = r2[0] * (1.0f / C_) - mean * mean;
  float rstd = rsqrtf(var + EPS_);
#pragma unroll
  for (int j = 0; j < 4; j++) {
    int cidx = tid + j * 128;
    float o = (v[j] - mean) * rstd * w[cidx] + bb[cidx];
    y1[(size_t)row * C_ + cidx]  = o;
    y1h[(size_t)row * C_ + cidx] = (f16)o;
  }
}

// ---------------------------------------------------------------------------
// 4) WMMA GEMM: out[M,N] = A[M,K] f16 @ Bt[N,K]^T f16 (+ bias), f32 or f16 out.
//    Macro-tile 128x128, K-step 32; 8 waves x (2 M-tiles x 4 N-tiles).
//    Double-buffered LDS filled by async global->LDS b128 copies (ASYNCcnt),
//    overlapped with WMMA on the other buffer.
// ---------------------------------------------------------------------------
__global__ void __launch_bounds__(256) gemm_f16_kernel(const f16* __restrict__ A,
                                                       const f16* __restrict__ Bt,
                                                       const float* __restrict__ bias,
                                                       float* __restrict__ outf,
                                                       f16* __restrict__ outh,
                                                       int M, int N, int K) {
  __shared__ __align__(16) f16 Ash[2][128][48];
  __shared__ __align__(16) f16 Bsh[2][128][48];
  int tid  = threadIdx.x;
  int wave = tid >> 5, lane = tid & 31;
  int li = lane & 15, kh = lane >> 4;
  int m0 = (wave & 3) * 32;        // wave's 32 rows (2 tiles)
  int n0 = (wave >> 2) * 64;       // wave's 64 cols (4 tiles)
  int mblock = blockIdx.y * 128;
  int nblock = blockIdx.x * 128;

  int lr = tid >> 2, lck = (tid & 3) * 8;        // slot 0: row lr, chunk lck
  int lr2 = lr + 64;                             // slot 1 (tid+256)

  auto issue = [&](int buf, int k0) {
    async_ld_b128(&Ash[buf][lr ][lck], A  + (size_t)(mblock + lr ) * K + k0 + lck);
    async_ld_b128(&Bsh[buf][lr ][lck], Bt + (size_t)(nblock + lr ) * K + k0 + lck);
    async_ld_b128(&Ash[buf][lr2][lck], A  + (size_t)(mblock + lr2) * K + k0 + lck);
    async_ld_b128(&Bsh[buf][lr2][lck], Bt + (size_t)(nblock + lr2) * K + k0 + lck);
  };

  v8f zero = {};
  v8f acc[2][4];
#pragma unroll
  for (int mi = 0; mi < 2; mi++)
#pragma unroll
    for (int ni = 0; ni < 4; ni++) acc[mi][ni] = zero;

  issue(0, 0);
  wait_async0();
  __syncthreads();

  int nk = K >> 5;
  for (int kt = 0; kt < nk; kt++) {
    int cur = kt & 1;
    if (kt + 1 < nk) issue(cur ^ 1, (kt + 1) * 32);

    v16h af[2], bf[4];
#pragma unroll
    for (int mi = 0; mi < 2; mi++) {
      int row = m0 + mi * 16 + li;
      v8h lo = *(const v8h*)&Ash[cur][row][kh * 8];
      v8h hi = *(const v8h*)&Ash[cur][row][16 + kh * 8];
      af[mi] = CAT16(lo, hi);
    }
#pragma unroll
    for (int ni = 0; ni < 4; ni++) {
      int row = n0 + ni * 16 + li;
      v8h lo = *(const v8h*)&Bsh[cur][row][kh * 8];
      v8h hi = *(const v8h*)&Bsh[cur][row][16 + kh * 8];
      bf[ni] = CAT16(lo, hi);
    }
#pragma unroll
    for (int mi = 0; mi < 2; mi++)
#pragma unroll
      for (int ni = 0; ni < 4; ni++)
        acc[mi][ni] = __builtin_amdgcn_wmma_f32_16x16x32_f16(false, af[mi], false, bf[ni],
                                                             (short)0, acc[mi][ni], false, false);
    wait_async0();
    __syncthreads();
  }

#pragma unroll
  for (int mi = 0; mi < 2; mi++) {
    int outm = mblock + m0 + mi * 16 + kh * 8;
#pragma unroll
    for (int ni = 0; ni < 4; ni++) {
      int n = nblock + n0 + ni * 16 + li;
      float bv = bias ? bias[n] : 0.0f;
      if (outh) {
#pragma unroll
        for (int r = 0; r < 8; r++)
          outh[(size_t)(outm + r) * N + n] = (f16)(acc[mi][ni][r] + bv);
      } else {
#pragma unroll
        for (int r = 0; r < 8; r++)
          outf[(size_t)(outm + r) * N + n] = acc[mi][ni][r] + bv;
      }
    }
  }
}

// ---------------------------------------------------------------------------
// 5) GLU gate: h = gelu(a) * sigmoid(gg), split from u [BT, 2H] f16; f16 out.
// ---------------------------------------------------------------------------
__global__ void gate_kernel(const f16* __restrict__ u, f16* __restrict__ hh) {
  int idx = blockIdx.x * blockDim.x + threadIdx.x;
  if (idx >= BT_ * C_) return;
  int row = idx >> 9, col = idx & (C_ - 1);
  float a  = (float)u[(size_t)row * NUP_ + col];
  float gg = (float)u[(size_t)row * NUP_ + C_ + col];
  float ge = 0.5f * a * (1.0f + erff(a * 0.70710678118654752f));
  float si = 1.0f / (1.0f + expf(-gg));
  hh[idx] = (f16)(ge * si);
}

// ---------------------------------------------------------------------------
// 6) Final: t1 = LN(y1 + z0, mix); out = LN(y1 + t1, ln2).  One block per row.
// ---------------------------------------------------------------------------
__global__ void __launch_bounds__(128) fuse2_kernel(const float* __restrict__ y1,
                                                    const float* __restrict__ z0,
                                                    const float* __restrict__ mw,
                                                    const float* __restrict__ mb,
                                                    const float* __restrict__ w2,
                                                    const float* __restrict__ b2,
                                                    float* __restrict__ outp) {
  __shared__ float r1[128], r2[128];
  int row = blockIdx.x, tid = threadIdx.x;
  float yv[4], v[4];
  float sum = 0.0f, sq = 0.0f;
#pragma unroll
  for (int j = 0; j < 4; j++) {
    int cidx = tid + j * 128;
    yv[j] = y1[(size_t)row * C_ + cidx];
    float vv = yv[j] + z0[(size_t)row * C_ + cidx];
    v[j] = vv; sum += vv; sq += vv * vv;
  }
  r1[tid] = sum; r2[tid] = sq; __syncthreads();
  for (int s = 64; s > 0; s >>= 1) {
    if (tid < s) { r1[tid] += r1[tid + s]; r2[tid] += r2[tid + s]; }
    __syncthreads();
  }
  float mean = r1[0] * (1.0f / C_);
  float var  = r2[0] * (1.0f / C_) - mean * mean;
  float rstd = rsqrtf(var + EPS_);
  sum = 0.0f; sq = 0.0f;
#pragma unroll
  for (int j = 0; j < 4; j++) {
    int cidx = tid + j * 128;
    float t1 = (v[j] - mean) * rstd * mw[cidx] + mb[cidx];
    float vv = yv[j] + t1;
    v[j] = vv; sum += vv; sq += vv * vv;
  }
  __syncthreads();
  r1[tid] = sum; r2[tid] = sq; __syncthreads();
  for (int s = 64; s > 0; s >>= 1) {
    if (tid < s) { r1[tid] += r1[tid + s]; r2[tid] += r2[tid + s]; }
    __syncthreads();
  }
  mean = r1[0] * (1.0f / C_);
  var  = r2[0] * (1.0f / C_) - mean * mean;
  rstd = rsqrtf(var + EPS_);
#pragma unroll
  for (int j = 0; j < 4; j++) {
    int cidx = tid + j * 128;
    outp[(size_t)row * C_ + cidx] = (v[j] - mean) * rstd * w2[cidx] + b2[cidx];
  }
}

// ---------------------------------------------------------------------------
extern "C" void kernel_launch(void* const* d_in, const int* in_sizes, int n_in,
                              void* d_out, int out_size, void* d_ws, size_t ws_size,
                              hipStream_t stream) {
  const float* x       = (const float*)d_in[0];
  const float* k       = (const float*)d_in[1];
  const float* se_w1   = (const float*)d_in[2];
  const float* se_b1   = (const float*)d_in[3];
  const float* se_w2   = (const float*)d_in[4];
  const float* se_b2   = (const float*)d_in[5];
  const float* ln1_w   = (const float*)d_in[6];
  const float* ln1_b   = (const float*)d_in[7];
  const float* up_w    = (const float*)d_in[8];
  const float* up_b    = (const float*)d_in[9];
  const float* down_w  = (const float*)d_in[10];
  const float* down_b  = (const float*)d_in[11];
  const float* mix_w   = (const float*)d_in[12];
  const float* mix_b   = (const float*)d_in[13];
  const float* ln2_w   = (const float*)d_in[14];
  const float* ln2_b   = (const float*)d_in[15];

  char* p = (char*)d_ws;
  auto alloc = [&](size_t bytes) -> void* {
    void* r = (void*)p;
    p += (bytes + 255) & ~(size_t)255;
    return r;
  };
  f16*   xT     = (f16*)alloc((size_t)C_ * BT_ * sizeof(f16));
  f16*   krev   = (f16*)alloc((size_t)C_ * L_ * sizeof(f16));
  f16*   upwT   = (f16*)alloc((size_t)NUP_ * C_ * sizeof(f16));  // [N,K]
  f16*   dwT    = (f16*)alloc((size_t)C_ * C_ * sizeof(f16));    // [N,K]
  float* ssum   = (float*)alloc((size_t)B_ * C_ * sizeof(float));
  float* g      = (float*)alloc((size_t)B_ * C_ * sizeof(float));
  f16*   y_h    = (f16*)alloc((size_t)BT_ * C_ * sizeof(f16));
  float* y1     = (float*)alloc((size_t)BT_ * C_ * sizeof(float));
  f16*   y1h    = (f16*)alloc((size_t)BT_ * C_ * sizeof(f16));
  f16*   u_h    = (f16*)alloc((size_t)BT_ * NUP_ * sizeof(f16));
  f16*   hh     = (f16*)alloc((size_t)BT_ * C_ * sizeof(f16));
  float* z0     = (float*)alloc((size_t)BT_ * C_ * sizeof(float));
  (void)ws_size; (void)n_in; (void)in_sizes; (void)out_size;

  // Stage 0: pack/convert operands for the WMMA paths
  pack_x_kernel<<<dim3(C_ / 32, T_ / 32, B_), dim3(32, 8), 0, stream>>>(x, xT);
  krev_kernel<<<(C_ * L_ + 255) / 256, 256, 0, stream>>>(k, krev);
  cvt_w_transpose_kernel<<<dim3(NUP_ / 32, C_ / 32), dim3(32, 8), 0, stream>>>(up_w, upwT, C_, NUP_);
  cvt_w_transpose_kernel<<<dim3(C_ / 32, C_ / 32), dim3(32, 8), 0, stream>>>(down_w, dwT, C_, C_);

  // Stage 1: block-Toeplitz WMMA depthwise conv (+ SE time-sums)
  conv_kernel<<<C_, 128, 0, stream>>>(xT, krev, y_h, ssum);

  // Stage 2: SE gate
  se_kernel<<<1, 256, 0, stream>>>(ssum, se_w1, se_b1, se_w2, se_b2, g);

  // Stage 3: y1 = LN(x + y*g)
  fuse1_kernel<<<BT_, 128, 0, stream>>>(x, y_h, g, ln1_w, ln1_b, y1, y1h);

  // Stage 4: up-projection GEMM (WMMA, async double-buffered, f16 out)
  gemm_f16_kernel<<<dim3(NUP_ / 128, BT_ / 128), 256, 0, stream>>>(y1h, upwT, up_b,
                                                                   nullptr, u_h,
                                                                   BT_, NUP_, C_);
  // Stage 5: GLU gating
  gate_kernel<<<(BT_ * C_ + 255) / 256, 256, 0, stream>>>(u_h, hh);

  // Stage 6: down-projection GEMM (WMMA, async double-buffered, f32 out)
  gemm_f16_kernel<<<dim3(C_ / 128, BT_ / 128), 256, 0, stream>>>(hh, dwT, down_b,
                                                                 z0, nullptr,
                                                                 BT_, C_, C_);
  // Stage 7: double LayerNorm -> output
  fuse2_kernel<<<BT_, 128, 0, stream>>>(y1, z0, mix_w, mix_b, ln2_w, ln2_b, (float*)d_out);
}
